// LocalTrittention_40286793236788
// MI455X (gfx1250) — matrix operand order repaired and checked by reading, not compiled, not run on hardware
//
#include <hip/hip_runtime.h>

// ---------------------------------------------------------------------------
// LocalTrittention for MI455X (gfx1250, wave32, WMMA 16x16x32 f16).
//
// Work: 4 fused projections (v = v1+v2) = 2.2 TFLOP dense GEMM (dominant),
// plus per-head (Q@K1^T)@K2^T softmax P@V = ~0.1 TFLOP.  Compute-bound =>
// everything runs on v_wmma_f32_16x16x32_f16 with f32 accumulate.
// All GEMMs are arranged as A @ Y^T with both operands row-major, so every
// WMMA A/B fragment is two contiguous 16B loads per lane (no transpose ops).
// V is stored transposed [B,H,D,S] by the projection epilogue to keep the
// P@V B-fragments contiguous too.
//
// The projection GEMM uses CDNA5 async Global->LDS copies (ASYNCcnt) with
// double-buffered LDS: the next 128x64 A/B tiles stream into LDS while the
// current tiles feed 16 WMMAs per wave; one s_wait_asynccnt + one barrier
// per K-chunk (vs load-stall + two barriers for the sync version).
//
// Workspace layout (assumes ws_size >= ~806 MB):
//   hs16   : [16384][4096] f16
//   w16    : [4][4096][4096] f16   (q, k1, k2, v1+v2)
//   b4     : [4][4096] f32
//   q16,k1_16,k2_16 : [B,H,S,D] f16 ;  vT16 : [B,H,D,S] f16
// ---------------------------------------------------------------------------

typedef __attribute__((ext_vector_type(16))) _Float16 v16h;
typedef __attribute__((ext_vector_type(8)))  _Float16 v8h;
typedef __attribute__((ext_vector_type(4)))  _Float16 v4h;
typedef __attribute__((ext_vector_type(8)))  float    v8f;
typedef __attribute__((ext_vector_type(4)))  int      int4v;

#if defined(__HIP_DEVICE_COMPILE__) &&                                   \
    __has_builtin(__builtin_amdgcn_global_load_async_to_lds_b128) &&     \
    __has_builtin(__builtin_amdgcn_s_wait_asynccnt)
#define USE_ASYNC_LDS 1
#else
#define USE_ASYNC_LDS 0
#endif

#define WMMA_F16(a, b, c) \
  __builtin_amdgcn_wmma_f32_16x16x32_f16(false, (a), false, (b), (short)0, (c), false, false)

__device__ __forceinline__ v16h cat16(v8h lo, v8h hi) {
  return __builtin_shufflevector(lo, hi, 0,1,2,3,4,5,6,7,8,9,10,11,12,13,14,15);
}

// A fragment (M=16 x K=32, 16-bit): lane r<16 holds row r, K chunks
// [k0 .. k0+7] and [k0+16 .. k0+23]; lanes 16..31 the +8 K-offset halves.
// p points at tile element [0][0]; ld = row stride in halves.
__device__ __forceinline__ v16h fragA(const _Float16* p, int ld, int lane) {
  const int r  = lane & 15;
  const int h8 = (lane >> 4) * 8;
  const _Float16* q = p + (size_t)r * ld + h8;
  v8h lo = *(const v8h*)(q);
  v8h hi = *(const v8h*)(q + 16);
  return cat16(lo, hi);
}

// B fragment for A@Y^T (Y row-major [N][K]): lane r<16 = column N=r,
// K = [k0..k0+7] and [k0+16..k0+23]; lanes 16..31 the +8 K-offset halves.
__device__ __forceinline__ v16h fragB(const _Float16* p, int ld, int lane) {
  const int r   = lane & 15;
  const int h16 = (lane >> 4) * 8;
  const _Float16* q = p + (size_t)r * ld + h16;
  v8h lo = *(const v8h*)(q);
  v8h hi = *(const v8h*)(q + 16);
  return cat16(lo, hi);
}

// ------------------------------- prep kernels ------------------------------

__global__ void cvt_f16_kernel(const float* __restrict__ in,
                               _Float16* __restrict__ out, int n4) {
  int i = blockIdx.x * blockDim.x + threadIdx.x;
  if (i < n4) {
    float4 v = ((const float4*)in)[i];
    v4h o = {(_Float16)v.x, (_Float16)v.y, (_Float16)v.z, (_Float16)v.w};
    ((v4h*)out)[i] = o;
  }
}

__global__ void prep_weights_kernel(const float* __restrict__ wq,
                                    const float* __restrict__ wk1,
                                    const float* __restrict__ wk2,
                                    const float* __restrict__ wv1,
                                    const float* __restrict__ wv2,
                                    _Float16* __restrict__ w16, int n4) {
  int i = blockIdx.x * blockDim.x + threadIdx.x;
  if (i >= n4) return;
  const size_t NW4 = (size_t)4096 * 4096 / 4;
  v4h* o = (v4h*)w16;
  float4 a = ((const float4*)wq)[i];
  o[i] = {(_Float16)a.x, (_Float16)a.y, (_Float16)a.z, (_Float16)a.w};
  float4 b = ((const float4*)wk1)[i];
  o[NW4 + i] = {(_Float16)b.x, (_Float16)b.y, (_Float16)b.z, (_Float16)b.w};
  float4 c = ((const float4*)wk2)[i];
  o[2 * NW4 + i] = {(_Float16)c.x, (_Float16)c.y, (_Float16)c.z, (_Float16)c.w};
  float4 d1 = ((const float4*)wv1)[i];
  float4 d2 = ((const float4*)wv2)[i];
  o[3 * NW4 + i] = {(_Float16)(d1.x + d2.x), (_Float16)(d1.y + d2.y),
                    (_Float16)(d1.z + d2.z), (_Float16)(d1.w + d2.w)};
}

__global__ void prep_bias_kernel(const float* __restrict__ bq,
                                 const float* __restrict__ bk1,
                                 const float* __restrict__ bk2,
                                 const float* __restrict__ bv1,
                                 const float* __restrict__ bv2,
                                 float* __restrict__ b4) {
  int i = blockIdx.x * blockDim.x + threadIdx.x;
  if (i < 4096) {
    b4[i]          = bq[i];
    b4[4096 + i]   = bk1[i];
    b4[8192 + i]   = bk2[i];
    b4[12288 + i]  = bv1[i] + bv2[i];
  }
}

// --------------------------- projection GEMM -------------------------------
// C[16384,4096] = hs16 @ W[z]^T + bias[z];  128x128 tile per block, 8 waves,
// each wave a 32x64 subtile (2x4 WMMA accumulators).  K chunk 64 via LDS,
// double-buffered with async Global->LDS fills.
// z<3 stores [B,H,S,D]; z==3 stores transposed [B,H,D,S].

__global__ __launch_bounds__(256) void proj_kernel(
    const _Float16* __restrict__ A,      // [16384][4096]
    const _Float16* __restrict__ Wall,   // [4][4096][4096]
    const float*    __restrict__ biasAll,// [4][4096]
    _Float16* __restrict__ Qo, _Float16* __restrict__ K1o,
    _Float16* __restrict__ K2o, _Float16* __restrict__ VTo) {
  constexpr int K   = 4096;
  constexpr int LDT = 72;  // 64 + 8 halves pad -> conflict-free b128 reads

  const int tid  = threadIdx.x;
  const int lane = tid & 31;
  const int w    = tid >> 5;
  const int z    = blockIdx.z;
  const int n0   = blockIdx.x * 128;
  const int m0   = blockIdx.y * 128;
  const _Float16* Wm = Wall + (size_t)z * K * K;
  const int mw = (w & 3) * 32;   // wave row offset in tile
  const int nw = (w >> 2) * 64;  // wave col offset in tile

  const int ldRow = tid >> 1;        // 0..127
  const int ldCol = (tid & 1) * 32;  // 0 / 32 halves (64B contiguous chunk)
  const _Float16* gaBase = A  + (size_t)(m0 + ldRow) * K + ldCol;
  const _Float16* gbBase = Wm + (size_t)(n0 + ldRow) * K + ldCol;

  v8f acc[2][4] = {};

#if USE_ASYNC_LDS
  __shared__ _Float16 As[2][128 * LDT];  // 2 x 18,432 B
  __shared__ _Float16 Bs[2][128 * LDT];  // 2 x 18,432 B   (73,728 B total)
  typedef __attribute__((address_space(1))) int4v gv4i;  // global int4*
  typedef __attribute__((address_space(3))) int4v lv4i;  // LDS int4*

  auto fill = [&](int buf, int kt) {
    const _Float16* ga = gaBase + kt;
    const _Float16* gb = gbBase + kt;
    _Float16* la = &As[buf][ldRow * LDT + ldCol];
    _Float16* lb = &Bs[buf][ldRow * LDT + ldCol];
    __builtin_amdgcn_global_load_async_to_lds_b128((gv4i*)ga, (lv4i*)la, 0, 0);
    __builtin_amdgcn_global_load_async_to_lds_b128((gv4i*)ga, (lv4i*)la, 16, 0);
    __builtin_amdgcn_global_load_async_to_lds_b128((gv4i*)ga, (lv4i*)la, 32, 0);
    __builtin_amdgcn_global_load_async_to_lds_b128((gv4i*)ga, (lv4i*)la, 48, 0);
    __builtin_amdgcn_global_load_async_to_lds_b128((gv4i*)gb, (lv4i*)lb, 0, 0);
    __builtin_amdgcn_global_load_async_to_lds_b128((gv4i*)gb, (lv4i*)lb, 16, 0);
    __builtin_amdgcn_global_load_async_to_lds_b128((gv4i*)gb, (lv4i*)lb, 32, 0);
    __builtin_amdgcn_global_load_async_to_lds_b128((gv4i*)gb, (lv4i*)lb, 48, 0);
  };

  fill(0, 0);
  __builtin_amdgcn_s_wait_asynccnt(0);
  __syncthreads();

  for (int kt = 0; kt < K; kt += 64) {
    const int cur = (kt >> 6) & 1;
    if (kt + 64 < K) fill(cur ^ 1, kt + 64);  // stream next tile during WMMAs
    const _Float16* as = As[cur];
    const _Float16* bs = Bs[cur];
#pragma unroll
    for (int kk = 0; kk < 64; kk += 32) {
      v16h af[2];
#pragma unroll
      for (int mi = 0; mi < 2; ++mi)
        af[mi] = fragA(as + (mw + mi * 16) * LDT + kk, LDT, lane);
#pragma unroll
      for (int ni = 0; ni < 4; ++ni) {
        v16h bf = fragB(bs + (nw + ni * 16) * LDT + kk, LDT, lane);
#pragma unroll
        for (int mi = 0; mi < 2; ++mi)
          acc[mi][ni] = WMMA_F16(af[mi], bf, acc[mi][ni]);
      }
    }
    __builtin_amdgcn_s_wait_asynccnt(0);
    __syncthreads();
  }
#else  // --------------- fallback: synchronous single-buffer path ----------
  __shared__ _Float16 As1[128 * LDT];
  __shared__ _Float16 Bs1[128 * LDT];
  for (int kt = 0; kt < K; kt += 64) {
    const _Float16* ga = gaBase + kt;
    const _Float16* gb = gbBase + kt;
    _Float16* la = As1 + ldRow * LDT + ldCol;
    _Float16* lb = Bs1 + ldRow * LDT + ldCol;
#pragma unroll
    for (int u = 0; u < 4; ++u) {
      *(v8h*)(la + u * 8) = *(const v8h*)(ga + u * 8);
      *(v8h*)(lb + u * 8) = *(const v8h*)(gb + u * 8);
    }
    __syncthreads();
#pragma unroll
    for (int kk = 0; kk < 64; kk += 32) {
      v16h af[2];
#pragma unroll
      for (int mi = 0; mi < 2; ++mi)
        af[mi] = fragA(As1 + (mw + mi * 16) * LDT + kk, LDT, lane);
#pragma unroll
      for (int ni = 0; ni < 4; ++ni) {
        v16h bf = fragB(Bs1 + (nw + ni * 16) * LDT + kk, LDT, lane);
#pragma unroll
        for (int mi = 0; mi < 2; ++mi)
          acc[mi][ni] = WMMA_F16(af[mi], bf, acc[mi][ni]);
      }
    }
    __syncthreads();
  }
#endif

  _Float16* dst = (z == 0) ? Qo : (z == 1) ? K1o : (z == 2) ? K2o : VTo;
  const bool tr = (z == 3);
#pragma unroll
  for (int mi = 0; mi < 2; ++mi) {
#pragma unroll
    for (int ni = 0; ni < 4; ++ni) {
      const int col = n0 + nw + ni * 16 + (lane & 15);
      const int hh = col >> 8, dd = col & 255;
      const float bias = biasAll[z * 4096 + col];
#pragma unroll
      for (int e = 0; e < 8; ++e) {
        const int row = m0 + mw + mi * 16 + e + ((lane >> 4) << 3);
        const int bb = row >> 8, ss = row & 255;
        const size_t idx =
            tr ? ((((size_t)bb * 16 + hh) * 256 + dd) * 256 + ss)
               : ((((size_t)bb * 16 + hh) * 256 + ss) * 256 + dd);
        dst[idx] = (_Float16)(acc[mi][ni][e] + bias);
      }
    }
  }
}

// ------------------------------ attention ----------------------------------
// One wave owns 32 query rows of one head; all stages wave-private.
// grid = (B*H, 4), block = 64 threads (2 waves, 2 * 32x264 f16 LDS strips).

__global__ __launch_bounds__(64) void attn_kernel(
    const _Float16* __restrict__ Q,  const _Float16* __restrict__ K1,
    const _Float16* __restrict__ K2, const _Float16* __restrict__ VT,
    float* __restrict__ Out) {
  constexpr int LD1 = 264;  // 256 + 8 halves pad (132 words % 64 == 4)
  __shared__ _Float16 sbuf[2 * 32 * LD1];  // 33.8 KB
  const int tid = threadIdx.x, lane = tid & 31, w = tid >> 5;
  const int bh = blockIdx.x;
  const int b = bh >> 4, h = bh & 15;
  const size_t hoff = (size_t)bh << 16;  // * 256*256
  const _Float16* q  = Q  + hoff;
  const _Float16* k1 = K1 + hoff;
  const _Float16* k2 = K2 + hoff;
  const _Float16* vt = VT + hoff;
  const int mw = blockIdx.y * 64 + w * 32;  // global query-row base
  _Float16* myb = sbuf + w * 32 * LD1;      // wave-private strip
  const int rlo = lane & 15;
  const int rhi = (lane >> 4) << 3;

  // ---- stage 1: s1 = Q @ K1^T  (32 x 256, f16 into LDS) ----
  for (int nt = 0; nt < 256; nt += 64) {
    v8f acc[2][4] = {};
    for (int k = 0; k < 256; k += 32) {
      v16h af[2];
#pragma unroll
      for (int mi = 0; mi < 2; ++mi)
        af[mi] = fragA(q + (size_t)(mw + mi * 16) * 256 + k, 256, lane);
#pragma unroll
      for (int ni = 0; ni < 4; ++ni) {
        v16h bf = fragB(k1 + (size_t)(nt + ni * 16) * 256 + k, 256, lane);
#pragma unroll
        for (int mi = 0; mi < 2; ++mi)
          acc[mi][ni] = WMMA_F16(af[mi], bf, acc[mi][ni]);
      }
    }
#pragma unroll
    for (int mi = 0; mi < 2; ++mi)
#pragma unroll
      for (int ni = 0; ni < 4; ++ni)
#pragma unroll
        for (int e = 0; e < 8; ++e)
          myb[(mi * 16 + e + rhi) * LD1 + nt + ni * 16 + rlo] =
              (_Float16)acc[mi][ni][e];
  }

  // ---- stage 2: scores = (s1 @ K2^T) * SCALE, only j < 192 (mask) ----
  v8f sc[2][12] = {};
  for (int k = 0; k < 256; k += 32) {
    v16h af[2];
#pragma unroll
    for (int mi = 0; mi < 2; ++mi)
      af[mi] = fragA(myb + (mi * 16) * LD1 + k, LD1, lane);
#pragma unroll
    for (int jt = 0; jt < 12; ++jt) {
      v16h bf = fragB(k2 + (size_t)(jt * 16) * 256 + k, 256, lane);
#pragma unroll
      for (int mi = 0; mi < 2; ++mi)
        sc[mi][jt] = WMMA_F16(af[mi], bf, sc[mi][jt]);
    }
  }

  // ---- softmax over j=0..191 (cols >=192 are -inf -> excluded) ----
  // Row r of a tile lives in one 16-lane half at element index e of each acc.
#pragma unroll
  for (int mi = 0; mi < 2; ++mi) {
#pragma unroll
    for (int e = 0; e < 8; ++e) {
      float m = -1e30f;
#pragma unroll
      for (int jt = 0; jt < 12; ++jt) {
        sc[mi][jt][e] *= 0.0625f;  // 1/sqrt(256)
        m = fmaxf(m, sc[mi][jt][e]);
      }
#pragma unroll
      for (int off = 1; off < 16; off <<= 1)
        m = fmaxf(m, __shfl_xor(m, off, 32));
      float ssum = 0.f;
#pragma unroll
      for (int jt = 0; jt < 12; ++jt) {
        float p = __expf(sc[mi][jt][e] - m);
        sc[mi][jt][e] = p;
        ssum += p;
      }
#pragma unroll
      for (int off = 1; off < 16; off <<= 1)
        ssum += __shfl_xor(ssum, off, 32);
      const float inv = 1.0f / ssum;
#pragma unroll
      for (int jt = 0; jt < 12; ++jt) sc[mi][jt][e] *= inv;
    }
  }

  // P (f16) back into the wave-private strip, cols 0..191
#pragma unroll
  for (int mi = 0; mi < 2; ++mi)
#pragma unroll
    for (int jt = 0; jt < 12; ++jt)
#pragma unroll
      for (int e = 0; e < 8; ++e)
        myb[(mi * 16 + e + rhi) * LD1 + jt * 16 + rlo] =
            (_Float16)sc[mi][jt][e];

  // ---- stage 3: ctx = P @ V  (contract j=0..191; V stored transposed) ----
  for (int dt = 0; dt < 256; dt += 64) {
    v8f acc[2][4] = {};
    for (int k = 0; k < 192; k += 32) {
      v16h af[2];
#pragma unroll
      for (int mi = 0; mi < 2; ++mi)
        af[mi] = fragA(myb + (mi * 16) * LD1 + k, LD1, lane);
#pragma unroll
      for (int ni = 0; ni < 4; ++ni) {
        v16h bf = fragB(vt + (size_t)(dt + ni * 16) * 256 + k, 256, lane);
#pragma unroll
        for (int mi = 0; mi < 2; ++mi)
          acc[mi][ni] = WMMA_F16(af[mi], bf, acc[mi][ni]);
      }
    }
#pragma unroll
    for (int mi = 0; mi < 2; ++mi)
#pragma unroll
      for (int ni = 0; ni < 4; ++ni)
#pragma unroll
        for (int e = 0; e < 8; ++e) {
          const int ss = mw + mi * 16 + e + rhi;
          const int dd = dt + ni * 16 + rlo;
          Out[(((size_t)b * 256 + ss) * 16 + h) * 256 + dd] = acc[mi][ni][e];
        }
  }
}

// ------------------------------- launcher ----------------------------------

extern "C" void kernel_launch(void* const* d_in, const int* in_sizes, int n_in,
                              void* d_out, int out_size, void* d_ws,
                              size_t ws_size, hipStream_t stream) {
  const float* hs  = (const float*)d_in[0];
  const float* wq  = (const float*)d_in[1];
  const float* bq  = (const float*)d_in[2];
  const float* wk1 = (const float*)d_in[3];
  const float* bk1 = (const float*)d_in[4];
  const float* wk2 = (const float*)d_in[5];
  const float* bk2 = (const float*)d_in[6];
  const float* wv1 = (const float*)d_in[7];
  const float* bv1 = (const float*)d_in[8];
  const float* wv2 = (const float*)d_in[9];
  const float* bv2 = (const float*)d_in[10];
  float* out = (float*)d_out;

  const size_t NHS = (size_t)16384 * 4096;  // B*S x HID
  const size_t NW  = (size_t)4096 * 4096;

  char* ws = (char*)d_ws;
  size_t off = 0;
  _Float16* hs16 = (_Float16*)(ws + off); off += NHS * 2;
  _Float16* w16  = (_Float16*)(ws + off); off += 4 * NW * 2;
  float*    b4   = (float*)(ws + off);    off += 4 * 4096 * 4;
  _Float16* q16  = (_Float16*)(ws + off); off += NHS * 2;
  _Float16* k116 = (_Float16*)(ws + off); off += NHS * 2;
  _Float16* k216 = (_Float16*)(ws + off); off += NHS * 2;
  _Float16* vT16 = (_Float16*)(ws + off); off += NHS * 2;
  (void)ws_size;

  const int n4hs = (int)(NHS / 4);  // 16,777,216
  cvt_f16_kernel<<<n4hs / 256, 256, 0, stream>>>(hs, hs16, n4hs);

  const int n4w = (int)(NW / 4);    // 4,194,304
  prep_weights_kernel<<<n4w / 256, 256, 0, stream>>>(wq, wk1, wk2, wv1, wv2,
                                                     w16, n4w);
  prep_bias_kernel<<<16, 256, 0, stream>>>(bq, bk1, bk2, bv1, bv2, b4);

  dim3 pgrid(4096 / 128, 16384 / 128, 4);  // (32, 128, 4)
  proj_kernel<<<pgrid, 256, 0, stream>>>(hs16, w16, b4, q16, k116, k216, vT16);

  dim3 agrid(64 * 16, 4);  // (B*H, S/64)
  attn_kernel<<<agrid, 64, 0, stream>>>(q16, k116, k216, vT16, out);
}
